// GNNThicknessPredictor_9070970929320
// MI455X (gfx1250) — compile-verified, matrix operand order checked by reading, not run on hardware
//
#include <hip/hip_runtime.h>

typedef __attribute__((ext_vector_type(16))) _Float16 v16h;
typedef __attribute__((ext_vector_type(8)))  float    v8f;

#define HDIM 128
#define FINC 16
#define OUTD 10
#define LN_EPS 1e-5f

// ---------------- WMMA fragment helpers (wave32, v_wmma_f32_16x16x32_f16) ----
// A-matrix (16x32 f16) per CDNA5 ISA 7.12.2:
//   lanes 0-15 : m=lane,    VGPR v<4 -> k=2v,2v+1 ; v>=4 -> k=16+2(v-4),..
//   lanes 16-31: m=lane-16, same with k offset +8
// Two contiguous 8-half runs per lane -> compiler emits 2x ds_load_b128.
__device__ __forceinline__ v16h wmma_load_a(const _Float16* s, int ld, int kbase) {
  const int lane = threadIdx.x & 31;
  const int m  = lane & 15;
  const int hs = (lane >> 4) << 3;   // 0 or 8
  v16h a;
#pragma unroll
  for (int v = 0; v < 8; ++v) {
    const int kb = (v < 4) ? (2 * v) : (16 + 2 * (v - 4));
    const int k  = kbase + kb + hs;
    a[2 * v]     = s[m * ld + k];
    a[2 * v + 1] = s[m * ld + k + 1];
  }
  return a;
}

// B fragments are pre-swizzled to [kstep][ntile][lane][16] in global memory:
// one contiguous 32B load per lane (2x global_load_b128), L2-resident.
__device__ __forceinline__ v16h wmma_load_b_swz(const _Float16* base, int tile_idx) {
  const int lane = threadIdx.x & 31;
  return *(const v16h*)(base + ((size_t)tile_idx * 32 + lane) * 16);
}

// ---------------- Weight pre-swizzle: (k,n) f32 -> fragment-order f16 --------
// out element o = [ks][nt][lane][slot]; slot=2*vg+hf maps to
// k = ks*32 + kb(vg) + hf + 8*(lane>=16), n = nt*16 + (lane&15).
// Element (k,n): k < Ksplit -> Wtop[k][n], else Wbot[k-Ksplit][n]; n>=Nvalid -> 0.
__global__ void swizzle_weights_kernel(const float* __restrict__ Wtop,
                                       const float* __restrict__ Wbot,
                                       _Float16* __restrict__ outp,
                                       int Ktot, int Ntot, int Ksplit, int Nvalid) {
  const int NT = Ntot >> 4;
  int o = blockIdx.x * blockDim.x + threadIdx.x;
  if (o >= Ktot * Ntot) return;
  int ks   = o / (NT * 512);
  int r    = o - ks * NT * 512;
  int nt   = r >> 9;
  int lane = (r >> 4) & 31;
  int slot = r & 15;
  int vg = slot >> 1, hf = slot & 1;
  int kb = (vg < 4) ? (2 * vg) : (16 + 2 * (vg - 4));
  int k  = ks * 32 + kb + hf + ((lane >> 4) << 3);
  int n  = nt * 16 + (lane & 15);
  float v = 0.f;
  if (n < Nvalid)
    v = (k < Ksplit) ? Wtop[k * Nvalid + n] : Wbot[(k - Ksplit) * Nvalid + n];
  outp[o] = (_Float16)v;
}

// ---------------- Edge kernels (memory-bound roofline term) ------------------
__global__ void degree_kernel(const int* __restrict__ dst, float* __restrict__ deg, int E) {
  int e = blockIdx.x * blockDim.x + threadIdx.x;
  if (e < E) atomicAdd(&deg[dst[e]], 1.0f);
}

__global__ void scatter_add_kernel(const float* __restrict__ h,
                                   const int* __restrict__ src, const int* __restrict__ dst,
                                   float* __restrict__ agg, int E, int C) {
  const int chunks = C >> 2;
  long long tid = (long long)blockIdx.x * blockDim.x + threadIdx.x;
  if (tid >= (long long)E * chunks) return;
  int e  = (int)(tid / chunks);
  int c4 = (int)(tid % chunks) * 4;
  int s = src[e], d = dst[e];
  const float4 v = *(const float4*)(h + (long long)s * C + c4);
  float* o = agg + (long long)d * C + c4;
  atomicAdd(o + 0, v.x); atomicAdd(o + 1, v.y);
  atomicAdd(o + 2, v.z); atomicAdd(o + 3, v.w);
}

// ---------------- Shared LN + ReLU epilogue ----------------------------------
__device__ __forceinline__ void ln_relu_store(const float* Cs, float* r1, float* r2,
                                              float* mu, float* rs,
                                              const float* lnw, const float* lnb,
                                              float* hout, int r0, int N, int tid) {
  // 256-thread two-level reduction: 8 elems/thread, then 16-wide finish.
  {
    int row = tid >> 4, seg = tid & 15;
    float s1 = 0.f, s2 = 0.f;
#pragma unroll
    for (int j = 0; j < 8; ++j) {
      float v = Cs[row * HDIM + seg * 8 + j];
      s1 += v; s2 += v * v;
    }
    r1[tid] = s1; r2[tid] = s2;
  }
  __syncthreads();
  if (tid < 16) {
    float a = 0.f, b = 0.f;
#pragma unroll
    for (int j = 0; j < 16; ++j) { a += r1[tid * 16 + j]; b += r2[tid * 16 + j]; }
    float m_ = a * (1.0f / HDIM);
    float v_ = fmaxf(b * (1.0f / HDIM) - m_ * m_, 0.f);
    mu[tid] = m_;
    rs[tid] = rsqrtf(v_ + LN_EPS);
  }
  __syncthreads();
  for (int i = tid; i < 16 * HDIM; i += 256) {
    int m = i >> 7, col = i & (HDIM - 1);
    int row = r0 + m;
    if (row < N) {
      float val = (Cs[i] - mu[m]) * rs[m] * lnw[col] + lnb[col];
      hout[(long long)row * HDIM + col] = fmaxf(val, 0.f);
    }
  }
}

// ---------------- SAGE layer 0: [mean(16)|x(16)] @ swz[Wl0;Wr0] --------------
__global__ __launch_bounds__(256) void sage_layer0_kernel(
    const float* __restrict__ x, const float* __restrict__ agg, const float* __restrict__ deg,
    const _Float16* __restrict__ Bswz, const float* __restrict__ bias,
    const float* __restrict__ lnw, const float* __restrict__ lnb,
    float* __restrict__ hout, int N)
{
  __shared__ __align__(16) _Float16 As[16 * 32];
  __shared__ float Cs[16 * HDIM];
  __shared__ float r1[256], r2[256], mu[16], rs[16];

  const int tid = threadIdx.x;
  const int r0  = blockIdx.x * 16;

  for (int i = tid; i < 16 * 32; i += 256) {
    int m = i >> 5, k = i & 31;
    int row = min(r0 + m, N - 1);
    float v = (k < FINC) ? agg[row * FINC + k] / fmaxf(deg[row], 1.0f)
                         : x[row * FINC + (k - FINC)];
    As[i] = (_Float16)v;
  }
  __syncthreads();

  const int wave = tid >> 5;
  const int n0 = wave * 16;
  v8f c = {};
  v16h a = wmma_load_a(As, 32, 0);
  v16h b = wmma_load_b_swz(Bswz, wave);        // single k-step, 8 n-tiles
  c = __builtin_amdgcn_wmma_f32_16x16x32_f16(false, a, false, b, (short)0, c, false, false);

  {
    const int lane = tid & 31;
    const int n = lane & 15, mh = (lane >> 4) << 3;
#pragma unroll
    for (int v = 0; v < 8; ++v)
      Cs[(v + mh) * HDIM + n0 + n] = c[v] + bias[n0 + n];
  }
  __syncthreads();
  ln_relu_store(Cs, r1, r2, mu, rs, lnw, lnb, hout, r0, N, tid);
}

// ---------------- SAGE layers 1/2: [mean(128)|h(128)] @ swz[Wl;Wr] -----------
__global__ __launch_bounds__(256) void sage_layer_kernel(
    const float* __restrict__ hin, const float* __restrict__ agg, const float* __restrict__ deg,
    const _Float16* __restrict__ Bswz,   // [8 ksteps][8 ntiles][32][16]
    const float* __restrict__ bias,
    const float* __restrict__ lnw, const float* __restrict__ lnb,
    float* __restrict__ hout, int N)
{
  __shared__ __align__(16) _Float16 As[16 * 256];
  __shared__ float Cs[16 * HDIM];
  __shared__ float r1[256], r2[256], mu[16], rs[16];

  const int tid = threadIdx.x;
  const int r0  = blockIdx.x * 16;

  for (int i = tid; i < 16 * 256; i += 256) {
    int m = i >> 8, k = i & 255;
    int row = min(r0 + m, N - 1);
    float v = (k < HDIM) ? agg[(long long)row * HDIM + k] / fmaxf(deg[row], 1.0f)
                         : hin[(long long)row * HDIM + (k - HDIM)];
    As[i] = (_Float16)v;
  }
  __syncthreads();

  const int wave = tid >> 5;
  const int n0 = wave * 16;
  v8f c = {};
#pragma unroll
  for (int ks = 0; ks < 8; ++ks) {          // barrier-free inner loop
    v16h a = wmma_load_a(As, 256, ks * 32);
    v16h b = wmma_load_b_swz(Bswz, ks * 8 + wave);
    c = __builtin_amdgcn_wmma_f32_16x16x32_f16(false, a, false, b, (short)0, c, false, false);
  }

  {
    const int lane = tid & 31;
    const int n = lane & 15, mh = (lane >> 4) << 3;
#pragma unroll
    for (int v = 0; v < 8; ++v)
      Cs[(v + mh) * HDIM + n0 + n] = c[v] + bias[n0 + n];
  }
  __syncthreads();
  ln_relu_store(Cs, r1, r2, mu, rs, lnw, lnb, hout, r0, N, tid);
}

// ---------------- Fused MLP: 128 -> 64 -> 32 -> 10, intermediates in LDS -----
__global__ __launch_bounds__(128) void mlp_kernel(
    const float* __restrict__ hin,
    const _Float16* __restrict__ w1s, const float* __restrict__ b1,
    const _Float16* __restrict__ w2s, const float* __restrict__ b2,
    const _Float16* __restrict__ w3s, const float* __restrict__ b3,
    float* __restrict__ out, int N)
{
  __shared__ __align__(16) _Float16 aS[16 * HDIM];
  __shared__ float oS[16 * 64];

  const int tid  = threadIdx.x;
  const int r0   = blockIdx.x * 16;
  const int wave = tid >> 5;
  const int lane = tid & 31;
  const int nn   = lane & 15;
  const int mh   = (lane >> 4) << 3;

  for (int i = tid; i < 16 * HDIM; i += 128) {
    int m = i >> 7, k = i & (HDIM - 1);
    int row = min(r0 + m, N - 1);
    aS[i] = (_Float16)hin[(long long)row * HDIM + k];
  }
  __syncthreads();

  // stage 1: [16x128] @ [128x64], 4 waves x 4 k-steps, w1s = [4][4][32][16]
  {
    v8f c = {};
#pragma unroll
    for (int ks = 0; ks < 4; ++ks) {
      v16h a = wmma_load_a(aS, HDIM, ks * 32);
      v16h b = wmma_load_b_swz(w1s, ks * 4 + wave);
      c = __builtin_amdgcn_wmma_f32_16x16x32_f16(false, a, false, b, (short)0, c, false, false);
    }
#pragma unroll
    for (int v = 0; v < 8; ++v)
      oS[(v + mh) * 64 + wave * 16 + nn] = c[v] + b1[wave * 16 + nn];
  }
  __syncthreads();
  for (int i = tid; i < 16 * 64; i += 128) aS[i] = (_Float16)fmaxf(oS[i], 0.f);
  __syncthreads();

  // stage 2: [16x64] @ [64x32], waves 0-1 (wave-uniform branch), w2s=[2][2][32][16]
  if (wave < 2) {
    v8f c = {};
#pragma unroll
    for (int ks = 0; ks < 2; ++ks) {
      v16h a = wmma_load_a(aS, 64, ks * 32);
      v16h b = wmma_load_b_swz(w2s, ks * 2 + wave);
      c = __builtin_amdgcn_wmma_f32_16x16x32_f16(false, a, false, b, (short)0, c, false, false);
    }
#pragma unroll
    for (int v = 0; v < 8; ++v)
      oS[(v + mh) * 32 + wave * 16 + nn] = c[v] + b2[wave * 16 + nn];
  }
  __syncthreads();
  for (int i = tid; i < 16 * 32; i += 128) aS[i] = (_Float16)fmaxf(oS[i], 0.f);
  __syncthreads();

  // stage 3: [16x32] @ [32x16(pad of 10)], wave 0 only, w3s=[1][1][32][16]
  if (wave == 0) {
    v8f c = {};
    v16h a = wmma_load_a(aS, 32, 0);
    v16h b = wmma_load_b_swz(w3s, 0);
    c = __builtin_amdgcn_wmma_f32_16x16x32_f16(false, a, false, b, (short)0, c, false, false);
    if (nn < OUTD) {
#pragma unroll
      for (int v = 0; v < 8; ++v) {
        int row = r0 + v + mh;
        if (row < N) out[(long long)row * OUTD + nn] = c[v] + b3[nn];
      }
    }
  }
}

// ---------------- Host orchestration -----------------------------------------
extern "C" void kernel_launch(void* const* d_in, const int* in_sizes, int n_in,
                              void* d_out, int out_size, void* d_ws, size_t ws_size,
                              hipStream_t stream)
{
  (void)n_in; (void)out_size; (void)ws_size;
  const float* x   = (const float*)d_in[0];
  const int*   ei  = (const int*)d_in[1];
  const float* Wl0 = (const float*)d_in[2];
  const float* Wr0 = (const float*)d_in[3];
  const float* bl0 = (const float*)d_in[4];
  const float* Wl  = (const float*)d_in[5];
  const float* Wr  = (const float*)d_in[6];
  const float* blv = (const float*)d_in[7];
  const float* lnw = (const float*)d_in[8];
  const float* lnb = (const float*)d_in[9];
  const float* W1  = (const float*)d_in[10];
  const float* b1  = (const float*)d_in[11];
  const float* W2  = (const float*)d_in[12];
  const float* b2  = (const float*)d_in[13];
  const float* W3  = (const float*)d_in[14];
  const float* b3  = (const float*)d_in[15];
  float* out = (float*)d_out;

  const int N = in_sizes[0] / FINC;
  const int E = in_sizes[1] / 2;
  const int* src = ei;
  const int* dst = ei + E;

  // ---- workspace layout (all offsets keep 32B alignment) ----
  float* deg = (float*)d_ws;                       // N
  float* agg = deg + N;                            // N*128
  float* h0  = agg + (size_t)N * HDIM;             // N*128
  float* h1  = h0  + (size_t)N * HDIM;             // N*128
  _Float16* wB0 = (_Float16*)(h1 + (size_t)N * HDIM); // 32*128
  _Float16* wB1 = wB0 + 32 * HDIM;                 // 256*128
  _Float16* wB2 = wB1 + 256 * HDIM;                // 256*128
  _Float16* wW1 = wB2 + 256 * HDIM;                // 128*64
  _Float16* wW2 = wW1 + HDIM * 64;                 // 64*32
  _Float16* wW3 = wW2 + 64 * 32;                   // 32*16

  const int nblk16 = (N + 15) / 16;

  // ---- pre-swizzle all weights into WMMA fragment order (tiny, L2-resident)
  swizzle_weights_kernel<<<(32 * 128 + 255) / 256, 256, 0, stream>>>(Wl0, Wr0, wB0, 32, 128, 16, 128);
  swizzle_weights_kernel<<<(256 * 128 + 255) / 256, 256, 0, stream>>>(Wl, Wr, wB1, 256, 128, 128, 128);
  swizzle_weights_kernel<<<(256 * 128 + 255) / 256, 256, 0, stream>>>(
      Wl + (size_t)HDIM * HDIM, Wr + (size_t)HDIM * HDIM, wB2, 256, 128, 128, 128);
  swizzle_weights_kernel<<<(128 * 64 + 255) / 256, 256, 0, stream>>>(W1, W1, wW1, 128, 64, 128, 64);
  swizzle_weights_kernel<<<(64 * 32 + 255) / 256, 256, 0, stream>>>(W2, W2, wW2, 64, 32, 64, 32);
  swizzle_weights_kernel<<<(32 * 16 + 255) / 256, 256, 0, stream>>>(W3, W3, wW3, 32, 16, 32, 10);

  hipMemsetAsync(deg, 0, (size_t)N * sizeof(float), stream);
  degree_kernel<<<(E + 255) / 256, 256, 0, stream>>>(dst, deg, E);

  // ---- layer 0
  hipMemsetAsync(agg, 0, (size_t)N * FINC * sizeof(float), stream);
  {
    long long work = (long long)E * (FINC / 4);
    scatter_add_kernel<<<(unsigned)((work + 255) / 256), 256, 0, stream>>>(x, src, dst, agg, E, FINC);
  }
  sage_layer0_kernel<<<nblk16, 256, 0, stream>>>(x, agg, deg, wB0, bl0, lnw, lnb, h0, N);

  // ---- layers 1..2
  const float* hin = h0;
  float* hout = h1;
  const _Float16* wB[2] = { wB1, wB2 };
  for (int l = 0; l < 2; ++l) {
    hipMemsetAsync(agg, 0, (size_t)N * HDIM * sizeof(float), stream);
    long long work = (long long)E * (HDIM / 4);
    scatter_add_kernel<<<(unsigned)((work + 255) / 256), 256, 0, stream>>>(hin, src, dst, agg, E, HDIM);
    sage_layer_kernel<<<nblk16, 256, 0, stream>>>(hin, agg, deg, wB[l],
        blv + (size_t)l * HDIM,
        lnw + (size_t)(l + 1) * HDIM, lnb + (size_t)(l + 1) * HDIM, hout, N);
    const float* t = hin; hin = hout; hout = const_cast<float*>(t);
  }

  // ---- fused regressor MLP
  mlp_kernel<<<nblk16, 128, 0, stream>>>(hin, wW1, b1, wW2, b2, wW3, b3, out, N);
}
